// SignalProp_16982300688962
// MI455X (gfx1250) — matrix operand order repaired and checked by reading, not compiled, not run on hardware
//
#include <hip/hip_runtime.h>
#include <stdint.h>

// ---------------------------------------------------------------------------
// SignalProp on MI455X (gfx1250): all MLP GEMMs via v_wmma_f32_16x16x32_f16.
// Block = 256 threads = 8 wave32. Hot kernels (netprop/lutq/luta) run two
// 16-row tiles per wave (256 rows/block) so each B fragment loaded from LDS
// feeds two WMMAs. Weights staged to LDS transposed as f16 ([N][K]) so B
// fragments are contiguous b32 reads (compiler merges to ds_load_b128).
// Leaky-ReLU as max(x, 0.2x): v_mul+v_max instead of cmp+cndmask.
// ---------------------------------------------------------------------------

typedef __attribute__((ext_vector_type(16))) _Float16 v16h;
typedef __attribute__((ext_vector_type(8)))  float    v8f;

#define LRELU 0.2f

__device__ __forceinline__ v8f wmma_f16(v16h a, v16h b, v8f c) {
  // 8 args: (neg_a, A, neg_b, B, c_mod, C, reuse_a, reuse_b)
  return __builtin_amdgcn_wmma_f32_16x16x32_f16(false, a, false, b, (short)0, c,
                                                false, false);
}

// A fragment (16x32 f16): lane m=lane%16, khalf=lane/16; VGPR r holds
// K = (r/4)*16 + khalf*8 + (r%4)*2 and K+1  (ISA 7.12.2).
__device__ __forceinline__ v16h load_A(const _Float16* tile, int stride, int lane, int kc) {
  union { v16h v; uint32_t u[8]; } a;
  int m = lane & 15, g = lane >> 4;
  const uint32_t* row = (const uint32_t*)(tile + m * stride + kc * 32);
#pragma unroll
  for (int r = 0; r < 8; ++r) {
    int k = ((r >> 2) << 4) + (g << 3) + ((r & 3) << 1);
    a.u[r] = row[k >> 1];
  }
  return a.v;
}

// B fragment (32x16 f16): lane n=lane%16, g=lane/16; VGPR r holds K=(r/4)*16+g*8+(r%4)*2,+1.
// wt is stored transposed in LDS: wt[n*Kp + k].
__device__ __forceinline__ v16h load_B(const _Float16* wt, int Kp, int lane, int kc, int nc) {
  union { v16h v; uint32_t u[8]; } b;
  int n = (lane & 15) + nc * 16, g = lane >> 4;
  const uint32_t* col = (const uint32_t*)(wt + n * Kp + kc * 32);
#pragma unroll
  for (int r = 0; r < 8; ++r) {
    int k = ((r >> 2) << 4) + (g << 3) + ((r & 3) << 1);
    b.u[r] = col[k >> 1];
  }
  return b.v;
}

__device__ __forceinline__ v8f bias_frag(const float* bs, int nc, int lane) {
  float bv = bs[nc * 16 + (lane & 15)];
  v8f c;
#pragma unroll
  for (int v = 0; v < 8; ++v) c[v] = bv;
  return c;
}

// D layout: VGPR v, lane: m = v + 8*(lane/16), n = lane%16 (+16*nc).
__device__ __forceinline__ void store_D(_Float16* tile, int stride, int lane, int nc,
                                        v8f c, bool lrelu) {
  int n = (lane & 15) + nc * 16, g = lane >> 4;
#pragma unroll
  for (int v = 0; v < 8; ++v) {
    float x = c[v];
    if (lrelu) x = fmaxf(x, LRELU * x);  // slope in (0,1): identical to leaky relu
    tile[(v + (g << 3)) * stride + n] = (_Float16)x;
  }
}

// 64 -> 64 layer, one 16-row tile (in-place; A frags fully in VGPRs before
// any store; DS ops of a wave are in order).
__device__ __forceinline__ void layer_64_64(_Float16* tile, int stride,
                                            const _Float16* wt, const float* bs,
                                            int lane, bool lrelu) {
  v16h a0 = load_A(tile, stride, lane, 0);
  v16h a1 = load_A(tile, stride, lane, 1);
#pragma unroll
  for (int nc = 0; nc < 4; ++nc) {
    v8f c = bias_frag(bs, nc, lane);
    c = wmma_f16(a0, load_B(wt, 64, lane, 0, nc), c);
    c = wmma_f16(a1, load_B(wt, 64, lane, 1, nc), c);
    store_D(tile, stride, lane, nc, c, lrelu);
  }
}

// 64 -> 64 layer over TWO 16-row tiles: B fragments reused across both tiles.
__device__ __forceinline__ void layer_64_64_x2(_Float16* tile, int stride,
                                               const _Float16* wt, const float* bs,
                                               int lane, bool lrelu) {
  _Float16* t1 = tile + 16 * stride;
  v16h a0 = load_A(tile, stride, lane, 0);
  v16h a1 = load_A(tile, stride, lane, 1);
  v16h e0 = load_A(t1, stride, lane, 0);
  v16h e1 = load_A(t1, stride, lane, 1);
#pragma unroll
  for (int nc = 0; nc < 4; ++nc) {
    v16h B0 = load_B(wt, 64, lane, 0, nc);
    v16h B1 = load_B(wt, 64, lane, 1, nc);
    v8f d0 = bias_frag(bs, nc, lane);
    d0 = wmma_f16(a0, B0, d0);
    d0 = wmma_f16(a1, B1, d0);
    v8f d1 = bias_frag(bs, nc, lane);
    d1 = wmma_f16(e0, B0, d1);
    d1 = wmma_f16(e1, B1, d1);
    store_D(tile, stride, lane, nc, d0, lrelu);
    store_D(t1, stride, lane, nc, d1, lrelu);
  }
}

// Kp=32 input layer -> 64, two tiles.
__device__ __forceinline__ void layer_32_64_x2(_Float16* tile, int stride,
                                               const _Float16* wt, const float* bs,
                                               int lane, bool lrelu) {
  _Float16* t1 = tile + 16 * stride;
  v16h a0 = load_A(tile, stride, lane, 0);
  v16h e0 = load_A(t1, stride, lane, 0);
#pragma unroll
  for (int nc = 0; nc < 4; ++nc) {
    v16h B0 = load_B(wt, 32, lane, 0, nc);
    v8f d0 = bias_frag(bs, nc, lane);
    d0 = wmma_f16(a0, B0, d0);
    v8f d1 = bias_frag(bs, nc, lane);
    d1 = wmma_f16(e0, B0, d1);
    store_D(tile, stride, lane, nc, d0, lrelu);
    store_D(t1, stride, lane, nc, d1, lrelu);
  }
}

// Global fp32 W[K][N] -> LDS f16 transposed wt[Np][Kp], zero padded.
__device__ __forceinline__ void stage_W(const float* W, int K, int N,
                                        _Float16* wt, int Kp, int Np, int tid, int nth) {
  for (int i = tid; i < Np * Kp; i += nth) {
    int n = i / Kp, k = i - n * Kp;
    wt[i] = (_Float16)((n < N && k < K) ? W[k * N + n] : 0.f);
  }
}
__device__ __forceinline__ void stage_b(const float* b, int N, float* bs, int Np,
                                        int tid, int nth) {
  for (int i = tid; i < Np; i += nth) bs[i] = (i < N) ? b[i] : 0.f;
}

__device__ __forceinline__ void atomicMaxF(float* addr, float val) {
  if (val >= 0.f) atomicMax((int*)addr, __float_as_int(val));
  else            atomicMin((unsigned int*)addr, __float_as_uint(val));
}

// ---------------------------------------------------------------------------
// init / PI scatter
// ---------------------------------------------------------------------------
__global__ __launch_bounds__(256) void k_init(float* pi_buf, float* net_sum,
                                              float* nfc1, float* nfc2,
                                              int* has_net, int* has_cell, int Nn) {
  int i = blockIdx.x * blockDim.x + threadIdx.x;
  if (i >= Nn) return;
  has_net[i] = 0; has_cell[i] = 0;
#pragma unroll
  for (int f = 0; f < 8; ++f) { pi_buf[(long)i * 8 + f] = 0.f; net_sum[(long)i * 8 + f] = 0.f; }
#pragma unroll
  for (int h = 0; h < 32; ++h) {
    nfc1[(long)i * 32 + h] = 0.f;
    nfc2[(long)i * 32 + h] = -3.4028234663852886e+38f;
  }
}

__global__ __launch_bounds__(256) void k_pi(const float* atslew, const int* pi,
                                            int NP, float* pi_buf) {
  int i = blockIdx.x * blockDim.x + threadIdx.x;
  if (i >= NP) return;
  int n = pi[i];
#pragma unroll
  for (int f = 0; f < 8; ++f) pi_buf[(long)n * 8 + f] = atslew[(long)n * 8 + f];
}

// ---------------------------------------------------------------------------
// netprop: [atslew[s](8), nf[s](10), nf[d](10)] -> 64x4 -> 8, scatter-add
// 256 rows/block, 2 tiles per wave.
// ---------------------------------------------------------------------------
__global__ __launch_bounds__(256) void k_netprop(
    const float* __restrict__ nf, const float* __restrict__ atslew,
    const int* __restrict__ src, const int* __restrict__ dst, int En,
    const float* W0, const float* b0, const float* W1, const float* b1,
    const float* W2, const float* b2, const float* W3, const float* b3,
    const float* W4, const float* b4,
    float* __restrict__ net_sum, int* __restrict__ has_net) {
  __shared__ _Float16 wt0[64 * 32];
  __shared__ _Float16 wt1[64 * 64];
  __shared__ _Float16 wt2[64 * 64];
  __shared__ _Float16 wt3[64 * 64];
  __shared__ _Float16 wt4[16 * 64];
  __shared__ float bs0[64], bs1[64], bs2[64], bs3[64], bs4[16];
  __shared__ _Float16 act[256 * 64];

  int tid = threadIdx.x;
  stage_W(W0, 28, 64, wt0, 32, 64, tid, 256);
  stage_W(W1, 64, 64, wt1, 64, 64, tid, 256);
  stage_W(W2, 64, 64, wt2, 64, 64, tid, 256);
  stage_W(W3, 64, 64, wt3, 64, 64, tid, 256);
  stage_W(W4, 64, 8, wt4, 64, 16, tid, 256);
  stage_b(b0, 64, bs0, 64, tid, 256); stage_b(b1, 64, bs1, 64, tid, 256);
  stage_b(b2, 64, bs2, 64, tid, 256); stage_b(b3, 64, bs3, 64, tid, 256);
  stage_b(b4, 8, bs4, 16, tid, 256);

  long base = (long)blockIdx.x * 256;
  {
    long e = base + tid;
    long ec = e < En ? e : (long)En - 1;
    int s = src[ec], d = dst[ec];
    _Float16* row = act + tid * 64;
#pragma unroll
    for (int i = 0; i < 8; ++i) row[i] = (_Float16)atslew[(long)s * 8 + i];
#pragma unroll
    for (int i = 0; i < 10; ++i) row[8 + i] = (_Float16)nf[(long)s * 10 + i];
#pragma unroll
    for (int i = 0; i < 10; ++i) row[18 + i] = (_Float16)nf[(long)d * 10 + i];
#pragma unroll
    for (int i = 28; i < 32; ++i) row[i] = (_Float16)0.f;
    if (e < En) has_net[d] = 1;
  }
  __syncthreads();

  int lane = tid & 31, w = tid >> 5;
  _Float16* tile = act + w * 32 * 64;

  layer_32_64_x2(tile, 64, wt0, bs0, lane, true);
  layer_64_64_x2(tile, 64, wt1, bs1, lane, true);
  layer_64_64_x2(tile, 64, wt2, bs2, lane, true);
  layer_64_64_x2(tile, 64, wt3, bs3, lane, true);

  // final 64 -> 8 (padded to 16), scatter-add straight from D fragments
  {
    v16h B0 = load_B(wt4, 64, lane, 0, 0);
    v16h B1 = load_B(wt4, 64, lane, 1, 0);
    int n = lane & 15, g = lane >> 4;
#pragma unroll
    for (int t = 0; t < 2; ++t) {
      _Float16* tt = tile + t * 16 * 64;
      v16h a0 = load_A(tt, 64, lane, 0);
      v16h a1 = load_A(tt, 64, lane, 1);
      v8f c = bias_frag(bs4, 0, lane);
      c = wmma_f16(a0, B0, c);
      c = wmma_f16(a1, B1, c);
      if (n < 8) {
#pragma unroll
        for (int v = 0; v < 8; ++v) {
          long e = base + (long)w * 32 + t * 16 + v + (g << 3);
          if (e < En) {
            int d = dst[e];
            atomicAdd(&net_sum[(long)d * 8 + n], c[v]);
          }
        }
      }
    }
  }
}

// ---------------------------------------------------------------------------
// lutq: [atslew[s], nf[s], nf[d]](28) -> 64x3 -> 64, store q per edge
// 256 rows/block, 2 tiles per wave.
// ---------------------------------------------------------------------------
__global__ __launch_bounds__(256) void k_lutq(
    const float* __restrict__ nf, const float* __restrict__ atslew,
    const int* __restrict__ src, const int* __restrict__ dst, int Ec,
    const float* W0, const float* b0, const float* W1, const float* b1,
    const float* W2, const float* b2, const float* W3, const float* b3,
    float* __restrict__ qbuf) {
  __shared__ _Float16 wt0[64 * 32];
  __shared__ _Float16 wt1[64 * 64];
  __shared__ _Float16 wt2[64 * 64];
  __shared__ _Float16 wt3[64 * 64];
  __shared__ float bs0[64], bs1[64], bs2[64], bs3[64];
  __shared__ _Float16 act[256 * 64];

  int tid = threadIdx.x;
  stage_W(W0, 28, 64, wt0, 32, 64, tid, 256);
  stage_W(W1, 64, 64, wt1, 64, 64, tid, 256);
  stage_W(W2, 64, 64, wt2, 64, 64, tid, 256);
  stage_W(W3, 64, 64, wt3, 64, 64, tid, 256);
  stage_b(b0, 64, bs0, 64, tid, 256); stage_b(b1, 64, bs1, 64, tid, 256);
  stage_b(b2, 64, bs2, 64, tid, 256); stage_b(b3, 64, bs3, 64, tid, 256);

  long base = (long)blockIdx.x * 256;
  {
    long e = base + tid;
    long ec = e < Ec ? e : (long)Ec - 1;
    int s = src[ec], d = dst[ec];
    _Float16* row = act + tid * 64;
#pragma unroll
    for (int i = 0; i < 8; ++i) row[i] = (_Float16)atslew[(long)s * 8 + i];
#pragma unroll
    for (int i = 0; i < 10; ++i) row[8 + i] = (_Float16)nf[(long)s * 10 + i];
#pragma unroll
    for (int i = 0; i < 10; ++i) row[18 + i] = (_Float16)nf[(long)d * 10 + i];
#pragma unroll
    for (int i = 28; i < 32; ++i) row[i] = (_Float16)0.f;
  }
  __syncthreads();

  int lane = tid & 31, w = tid >> 5;
  _Float16* tile = act + w * 32 * 64;
  layer_32_64_x2(tile, 64, wt0, bs0, lane, true);
  layer_64_64_x2(tile, 64, wt1, bs1, lane, true);
  layer_64_64_x2(tile, 64, wt2, bs2, lane, true);
  layer_64_64_x2(tile, 64, wt3, bs3, lane, false);
  __syncthreads();

  {
    long e = base + tid;
    if (e < Ec) {
      const _Float16* row = act + tid * 64;
#pragma unroll
      for (int i = 0; i < 64; ++i) qbuf[e * 64 + i] = (float)row[i];
    }
  }
}

// ---------------------------------------------------------------------------
// luta: 960k rows [q(2), axis(15)] -> 64x3 -> 14; outer(7x7) . tables -> r
// 256 rows/block, 2 tiles per wave. Dominant kernel (~19.5 GFLOP).
// ---------------------------------------------------------------------------
__global__ __launch_bounds__(256) void k_luta(
    const float* __restrict__ qbuf, const float* __restrict__ ef_cell, int Ec,
    const float* W0, const float* b0, const float* W1, const float* b1,
    const float* W2, const float* b2, const float* W3, const float* b3,
    float* __restrict__ rbuf) {
  __shared__ _Float16 wt0[64 * 32];
  __shared__ _Float16 wt1[64 * 64];
  __shared__ _Float16 wt2[64 * 64];
  __shared__ _Float16 wt3[16 * 64];
  __shared__ float bs0[64], bs1[64], bs2[64], bs3[16];
  __shared__ _Float16 act[256 * 64];

  int tid = threadIdx.x;
  stage_W(W0, 17, 64, wt0, 32, 64, tid, 256);
  stage_W(W1, 64, 64, wt1, 64, 64, tid, 256);
  stage_W(W2, 64, 64, wt2, 64, 64, tid, 256);
  stage_W(W3, 64, 14, wt3, 64, 16, tid, 256);
  stage_b(b0, 64, bs0, 64, tid, 256); stage_b(b1, 64, bs1, 64, tid, 256);
  stage_b(b2, 64, bs2, 64, tid, 256); stage_b(b3, 14, bs3, 16, tid, 256);

  long M = (long)Ec * 32;
  long base = (long)blockIdx.x * 256;
  {
    long j = base + tid;
    long jc = j < M ? j : M - 1;
    int e = (int)(jc >> 5), ld = (int)(jc & 31), l = ld >> 2;
    _Float16* row = act + tid * 64;
    row[0] = (_Float16)qbuf[(long)e * 64 + ld * 2 + 0];
    row[1] = (_Float16)qbuf[(long)e * 64 + ld * 2 + 1];
    const float* axis = ef_cell + (long)e * 512 + l * 15;
#pragma unroll
    for (int i = 0; i < 15; ++i) row[2 + i] = (_Float16)axis[i];
#pragma unroll
    for (int i = 17; i < 32; ++i) row[i] = (_Float16)0.f;
  }
  __syncthreads();

  int lane = tid & 31, w = tid >> 5;
  _Float16* tile = act + w * 32 * 64;
  layer_32_64_x2(tile, 64, wt0, bs0, lane, true);
  layer_64_64_x2(tile, 64, wt1, bs1, lane, true);
  layer_64_64_x2(tile, 64, wt2, bs2, lane, true);
  // final 64 -> 14 (one padded N chunk), B reused across both tiles
  {
    v16h B0 = load_B(wt3, 64, lane, 0, 0);
    v16h B1 = load_B(wt3, 64, lane, 1, 0);
#pragma unroll
    for (int t = 0; t < 2; ++t) {
      _Float16* tt = tile + t * 16 * 64;
      v16h a0 = load_A(tt, 64, lane, 0);
      v16h a1 = load_A(tt, 64, lane, 1);
      v8f c = bias_frag(bs3, 0, lane);
      c = wmma_f16(a0, B0, c);
      c = wmma_f16(a1, B1, c);
      store_D(tt, 64, lane, 0, c, false);
    }
  }
  __syncthreads();

  {
    long j = base + tid;
    if (j < M) {
      const _Float16* o = act + tid * 64;
      int e = (int)(j >> 5), ld = (int)(j & 31), l = ld >> 2;
      const float* tab = ef_cell + (long)e * 512 + 120 + l * 49;
      float r = 0.f;
#pragma unroll
      for (int i = 0; i < 7; ++i) {
        float a0v = (float)o[i];
#pragma unroll
        for (int jj = 0; jj < 7; ++jj) r += a0v * (float)o[7 + jj] * tab[i * 7 + jj];
      }
      rbuf[(long)e * 32 + ld] = r;
    }
  }
}

// ---------------------------------------------------------------------------
// cellarc: [last(8), nf[s](10), nf[d](10), r(32)](60) -> 64x3 -> 65
//          k=sigmoid(xc0); scatter-add efc1, scatter-max efc2
// 128 rows/block (80-wide tiles; LDS budget).
// ---------------------------------------------------------------------------
__global__ __launch_bounds__(256) void k_cellarc(
    const float* __restrict__ nf, const float* __restrict__ atslew,
    const int* __restrict__ src, const int* __restrict__ dst, int Ec,
    const float* __restrict__ rbuf,
    const float* W0, const float* b0, const float* W1, const float* b1,
    const float* W2, const float* b2, const float* W3, const float* b3,
    float* __restrict__ nfc1, float* __restrict__ nfc2, int* __restrict__ has_cell) {
  __shared__ _Float16 wt0[64 * 64];
  __shared__ _Float16 wt1[64 * 64];
  __shared__ _Float16 wt2[64 * 64];
  __shared__ _Float16 wt3[80 * 64];
  __shared__ float bs0[64], bs1[64], bs2[64], bs3[80];
  __shared__ _Float16 act[128 * 80];

  int tid = threadIdx.x;
  stage_W(W0, 60, 64, wt0, 64, 64, tid, 256);
  stage_W(W1, 64, 64, wt1, 64, 64, tid, 256);
  stage_W(W2, 64, 64, wt2, 64, 64, tid, 256);
  stage_W(W3, 64, 65, wt3, 64, 80, tid, 256);
  stage_b(b0, 64, bs0, 64, tid, 256); stage_b(b1, 64, bs1, 64, tid, 256);
  stage_b(b2, 64, bs2, 64, tid, 256); stage_b(b3, 65, bs3, 80, tid, 256);

  long base = (long)blockIdx.x * 128;
  if (tid < 128) {
    long e = base + tid;
    long ec = e < Ec ? e : (long)Ec - 1;
    int s = src[ec], d = dst[ec];
    _Float16* row = act + tid * 80;
#pragma unroll
    for (int i = 0; i < 8; ++i) row[i] = (_Float16)atslew[(long)s * 8 + i];
#pragma unroll
    for (int i = 0; i < 10; ++i) row[8 + i] = (_Float16)nf[(long)s * 10 + i];
#pragma unroll
    for (int i = 0; i < 10; ++i) row[18 + i] = (_Float16)nf[(long)d * 10 + i];
#pragma unroll
    for (int i = 0; i < 32; ++i) row[28 + i] = (_Float16)rbuf[ec * 32 + i];
#pragma unroll
    for (int i = 60; i < 64; ++i) row[i] = (_Float16)0.f;
  }
  __syncthreads();

  int lane = tid & 31, w = tid >> 5;
  _Float16* tile = act + w * 16 * 80;
  layer_64_64(tile, 80, wt0, bs0, lane, true);
  layer_64_64(tile, 80, wt1, bs1, lane, true);
  layer_64_64(tile, 80, wt2, bs2, lane, true);
  // final 64 -> 65 (padded to 80 = 5 N chunks)
  {
    v16h a0 = load_A(tile, 80, lane, 0);
    v16h a1 = load_A(tile, 80, lane, 1);
#pragma unroll
    for (int nc = 0; nc < 5; ++nc) {
      v8f c = bias_frag(bs3, nc, lane);
      c = wmma_f16(a0, load_B(wt3, 64, lane, 0, nc), c);
      c = wmma_f16(a1, load_B(wt3, 64, lane, 1, nc), c);
      store_D(tile, 80, lane, nc, c, false);
    }
  }
  __syncthreads();

  if (tid < 128) {
    long e = base + tid;
    if (e < Ec) {
      const _Float16* xc = act + tid * 80;
      float kk = 1.f / (1.f + __expf(-(float)xc[0]));
      int d = dst[e];
      has_cell[d] = 1;
#pragma unroll
      for (int h = 0; h < 32; ++h) {
        atomicAdd(&nfc1[(long)d * 32 + h], (float)xc[1 + h] * kk);
        atomicMaxF(&nfc2[(long)d * 32 + h], (float)xc[33 + h] * kk);
      }
    }
  }
}

// ---------------------------------------------------------------------------
// cellred + final select: [nf(10), nfc1(32), nfc2(32)](74->96) -> 64x3 -> 8
// out = has_cell ? cellred : has_net ? net_sum : pi_buf
// 128 rows/block (96-wide tiles; LDS budget).
// ---------------------------------------------------------------------------
__global__ __launch_bounds__(256) void k_cellred(
    const float* __restrict__ nf, const float* __restrict__ nfc1,
    const float* __restrict__ nfc2, const int* __restrict__ has_cell,
    const int* __restrict__ has_net, const float* __restrict__ net_sum,
    const float* __restrict__ pi_buf, int Nn,
    const float* W0, const float* b0, const float* W1, const float* b1,
    const float* W2, const float* b2, const float* W3, const float* b3,
    float* __restrict__ out) {
  __shared__ _Float16 wt0[64 * 96];
  __shared__ _Float16 wt1[64 * 64];
  __shared__ _Float16 wt2[64 * 64];
  __shared__ _Float16 wt3[16 * 64];
  __shared__ float bs0[64], bs1[64], bs2[64], bs3[16];
  __shared__ _Float16 act[128 * 96];

  int tid = threadIdx.x;
  stage_W(W0, 74, 64, wt0, 96, 64, tid, 256);
  stage_W(W1, 64, 64, wt1, 64, 64, tid, 256);
  stage_W(W2, 64, 64, wt2, 64, 64, tid, 256);
  stage_W(W3, 64, 8, wt3, 64, 16, tid, 256);
  stage_b(b0, 64, bs0, 64, tid, 256); stage_b(b1, 64, bs1, 64, tid, 256);
  stage_b(b2, 64, bs2, 64, tid, 256); stage_b(b3, 8, bs3, 16, tid, 256);

  long base = (long)blockIdx.x * 128;
  if (tid < 128) {
    long i = base + tid;
    long ic = i < Nn ? i : (long)Nn - 1;
    int hc = has_cell[ic];
    _Float16* row = act + tid * 96;
#pragma unroll
    for (int f = 0; f < 10; ++f) row[f] = (_Float16)nf[ic * 10 + f];
#pragma unroll
    for (int h = 0; h < 32; ++h) row[10 + h] = (_Float16)nfc1[ic * 32 + h];
#pragma unroll
    for (int h = 0; h < 32; ++h) row[42 + h] = (_Float16)(hc ? nfc2[ic * 32 + h] : 0.f);
#pragma unroll
    for (int f = 74; f < 96; ++f) row[f] = (_Float16)0.f;
  }
  __syncthreads();

  int lane = tid & 31, w = tid >> 5;
  _Float16* tile = act + w * 16 * 96;
  // layer0: K=96 (3 chunks) -> 64
  {
    v16h a0 = load_A(tile, 96, lane, 0);
    v16h a1 = load_A(tile, 96, lane, 1);
    v16h a2 = load_A(tile, 96, lane, 2);
#pragma unroll
    for (int nc = 0; nc < 4; ++nc) {
      v8f c = bias_frag(bs0, nc, lane);
      c = wmma_f16(a0, load_B(wt0, 96, lane, 0, nc), c);
      c = wmma_f16(a1, load_B(wt0, 96, lane, 1, nc), c);
      c = wmma_f16(a2, load_B(wt0, 96, lane, 2, nc), c);
      store_D(tile, 96, lane, nc, c, true);
    }
  }
  layer_64_64(tile, 96, wt1, bs1, lane, true);
  layer_64_64(tile, 96, wt2, bs2, lane, true);
  // final 64 -> 8 (padded N chunk)
  {
    v16h a0 = load_A(tile, 96, lane, 0);
    v16h a1 = load_A(tile, 96, lane, 1);
    v8f c = bias_frag(bs3, 0, lane);
    c = wmma_f16(a0, load_B(wt3, 64, lane, 0, 0), c);
    c = wmma_f16(a1, load_B(wt3, 64, lane, 1, 0), c);
    store_D(tile, 96, lane, 0, c, false);
  }
  __syncthreads();

  if (tid < 128) {
    long i = base + tid;
    if (i < Nn) {
      const _Float16* o = act + tid * 96;
      bool hc = has_cell[i] != 0, hn = has_net[i] != 0;
#pragma unroll
      for (int f = 0; f < 8; ++f) {
        float v;
        if (hc)      v = (float)o[f];
        else if (hn) v = net_sum[i * 8 + f];
        else         v = pi_buf[i * 8 + f];
        out[i * 8 + f] = v;
      }
    }
  }
}

// ---------------------------------------------------------------------------
extern "C" void kernel_launch(void* const* d_in, const int* in_sizes, int n_in,
                              void* d_out, int out_size, void* d_ws, size_t ws_size,
                              hipStream_t stream) {
  const float* nf      = (const float*)d_in[0];
  const float* atslew  = (const float*)d_in[1];
  const float* ef_cell = (const float*)d_in[2];
  const int* net_src   = (const int*)d_in[3];
  const int* net_dst   = (const int*)d_in[4];
  const int* cell_src  = (const int*)d_in[5];
  const int* cell_dst  = (const int*)d_in[6];
  const int* pi_nodes  = (const int*)d_in[7];

  int Nn  = in_sizes[0] / 10;
  int En  = in_sizes[3];
  int Ec  = in_sizes[5];
  int NPi = in_sizes[7];

  // Param leaves in dict order: netprop(10), lutq(8), luta(8), cellarc(8), cellred(8)
  const float* P[42];
  for (int i = 0; i < 42; ++i) P[i] = (const float*)d_in[8 + i];
  const float** np = P;        // netprop W0,b0,...,W4,b4
  const float** lq = P + 10;   // lutq
  const float** la = P + 18;   // luta
  const float** ca = P + 26;   // cellarc
  const float** cr = P + 34;   // cellred

  float* ws = (float*)d_ws;
  size_t off = 0;
  float* pi_buf  = ws + off; off += (size_t)Nn * 8;
  float* net_sum = ws + off; off += (size_t)Nn * 8;
  float* nfc1    = ws + off; off += (size_t)Nn * 32;
  float* nfc2    = ws + off; off += (size_t)Nn * 32;
  float* qbuf    = ws + off; off += (size_t)Ec * 64;
  float* rbuf    = ws + off; off += (size_t)Ec * 32;
  int* has_net   = (int*)(ws + off); off += (size_t)Nn;
  int* has_cell  = (int*)(ws + off); off += (size_t)Nn;
  (void)ws_size; (void)out_size; (void)n_in;

  k_init<<<(Nn + 255) / 256, 256, 0, stream>>>(pi_buf, net_sum, nfc1, nfc2,
                                               has_net, has_cell, Nn);
  k_pi<<<(NPi + 255) / 256, 256, 0, stream>>>(atslew, pi_nodes, NPi, pi_buf);

  k_netprop<<<(En + 255) / 256, 256, 0, stream>>>(
      nf, atslew, net_src, net_dst, En,
      np[0], np[1], np[2], np[3], np[4], np[5], np[6], np[7], np[8], np[9],
      net_sum, has_net);

  k_lutq<<<(Ec + 255) / 256, 256, 0, stream>>>(
      nf, atslew, cell_src, cell_dst, Ec,
      lq[0], lq[1], lq[2], lq[3], lq[4], lq[5], lq[6], lq[7], qbuf);

  long M = (long)Ec * 32;
  k_luta<<<(int)((M + 255) / 256), 256, 0, stream>>>(
      qbuf, ef_cell, Ec,
      la[0], la[1], la[2], la[3], la[4], la[5], la[6], la[7], rbuf);

  k_cellarc<<<(Ec + 127) / 128, 256, 0, stream>>>(
      nf, atslew, cell_src, cell_dst, Ec, rbuf,
      ca[0], ca[1], ca[2], ca[3], ca[4], ca[5], ca[6], ca[7],
      nfc1, nfc2, has_cell);

  k_cellred<<<(Nn + 127) / 128, 256, 0, stream>>>(
      nf, nfc1, nfc2, has_cell, has_net, net_sum, pi_buf, Nn,
      cr[0], cr[1], cr[2], cr[3], cr[4], cr[5], cr[6], cr[7],
      (float*)d_out);
}
